// GQAAttention_38912403702355
// MI455X (gfx1250) — matrix operand order repaired and checked by reading, not compiled, not run on hardware
//
#include <hip/hip_runtime.h>
#include <stdint.h>

// ---------------------------------------------------------------------------
// GQA attention layer for MI455X (gfx1250). All matmuls on
// v_wmma_f32_16x16x32_bf16 (fp32 accumulate); softmax/RoPE/accum in fp32.
// ~77 GFLOP over ~100MB traffic -> memory floor ~4.3us @ 23.3 TB/s.
//
// v4: GEMM staging via CDNA5 async copies (global_load_async_to_lds_b128,
// ASYNCcnt-gated) with double-buffered LDS; fragments pre-loaded into
// distinct registers so WMMAs overlap ds_load latency (staged dscnt waits);
// t+2 tile prefetch warms L2 ahead of the async engine.
// ---------------------------------------------------------------------------

typedef __attribute__((ext_vector_type(16))) __bf16   v16bf;
typedef __attribute__((ext_vector_type(8)))  float    v8f;
typedef __attribute__((ext_vector_type(8)))  uint32_t v8u;
typedef __attribute__((ext_vector_type(4)))  uint32_t v4u;

#define S_LEN 2048
#define DMODEL 2048
#define NHEAD 32
#define NKVH 8
#define HEADD 64

static __device__ __forceinline__ v16bf combine_bf(v4u lo, v4u hi) {
    v8u u;
    for (int i = 0; i < 4; ++i) { u[i] = lo[i]; u[i + 4] = hi[i]; }
    return __builtin_bit_cast(v16bf, u);
}

// A-matrix fragment (16x32 bf16) from a row-major 32-element row slice.
// lanes 0-15 hold K=0-15, lanes 16-31 hold K=16-31 (matches B fragment);
// per lane: 16 contiguous halves [2*kb, 2*kb+16), kb = 8*(lane>>4).
static __device__ __forceinline__ v16bf load_a_frag(const __bf16* rowbase,
                                                    int kb) {
    v4u lo = *(const v4u*)(rowbase + 2 * kb);
    v4u hi = *(const v4u*)(rowbase + 2 * kb + 8);
    return combine_bf(lo, hi);
}

// B-matrix fragment (32x16 bf16) column slice from K-contiguous storage:
// lane group selects K-half (grp*16); per lane 16 contiguous halves.
static __device__ __forceinline__ v16bf load_b_frag(const __bf16* colbase,
                                                    int grp) {
    v4u lo = *(const v4u*)(colbase + grp * 16);
    v4u hi = *(const v4u*)(colbase + grp * 16 + 8);
    return combine_bf(lo, hi);
}

// Async 16B x2 copy: global -> LDS (IOFFSET applies to both addresses).
static __device__ __forceinline__ void async_copy_32B(uint32_t lds_off,
                                                      const __bf16* gptr) {
    uint64_t ga = (uint64_t)(uintptr_t)gptr;
    asm volatile("global_load_async_to_lds_b128 %0, %1, off"
                 :: "v"(lds_off), "v"(ga) : "memory");
    asm volatile("global_load_async_to_lds_b128 %0, %1, off offset:16"
                 :: "v"(lds_off), "v"(ga) : "memory");
}

// ---------------------------------------------------------------------------
// fp32 -> bf16 elementwise convert (for x)
// ---------------------------------------------------------------------------
__global__ void cvt_f32_bf16(const float* __restrict__ in,
                             __bf16* __restrict__ out, int n) {
    int i = blockIdx.x * blockDim.x + threadIdx.x;
    if (i < n) out[i] = (__bf16)in[i];
}

// ---------------------------------------------------------------------------
// fp32 [R][C] -> bf16 transposed [C][R], 32x32 LDS tile (coalesced both ways)
// ---------------------------------------------------------------------------
__global__ __launch_bounds__(256) void cvt_transpose_bf16(
    const float* __restrict__ in, __bf16* __restrict__ out, int R, int C) {
    __shared__ float t[32][33];
    const int tx = threadIdx.x, ty = threadIdx.y;
    const int gx = blockIdx.x * 32;  // C direction
    const int gy = blockIdx.y * 32;  // R direction
    for (int r = ty; r < 32; r += 8)
        t[r][tx] = in[(size_t)(gy + r) * C + gx + tx];
    __syncthreads();
    for (int r = ty; r < 32; r += 8)
        out[(size_t)(gx + r) * R + gy + tx] = (__bf16)t[tx][r];
}

// ---------------------------------------------------------------------------
// bf16 WMMA GEMM with async double-buffered LDS staging:
//   C[M,N] = X[M,K] @ W[K,N], W given TRANSPOSED as Wt[N][K].
// Block = 128 threads (4 waves), tile 64(M)x64(N), K-step 32.
// Epilogue: optional RoPE (64-col tile == one head), fp32 out, bf16 out
// (optionally transposed for the attention V operand), bf16 scale.
// ---------------------------------------------------------------------------
__global__ __launch_bounds__(128) void gemm_bf16(
    const __bf16* __restrict__ X, const __bf16* __restrict__ Wt,
    float* __restrict__ outF, __bf16* __restrict__ outB,
    const float* __restrict__ cosT, const float* __restrict__ sinT,
    int M, int N, int K, int doRope, int transB, float bscale) {
    __shared__ __bf16 As[2][64 * 32];  // [m][k] row-major, double buffered
    __shared__ __bf16 Bs[2][64 * 32];  // [n][k] row-major, double buffered

    const int tid  = threadIdx.x;
    const int w    = tid >> 5;
    const int lane = tid & 31;
    const int nr   = lane & 15;
    const int grp  = lane >> 4;
    const int kb   = grp * 8;
    const int m0   = blockIdx.y * 64;
    const int n0   = blockIdx.x * 64;

    // Staging map: 64 rows x 32 bf16 = 4KB per tile; 32B chunk per thread.
    const int srow = tid >> 1;         // tile row (0..63)
    const int sc4  = (tid & 1) * 16;   // bf16 column of 32B chunk

    // LDS byte offsets of this thread's staging chunk (flat addr low 32 bits
    // == workgroup LDS offset per ISA 10.2 aperture mapping).
    uint32_t ldsA[2], ldsB[2];
    for (int b = 0; b < 2; ++b) {
        ldsA[b] = (uint32_t)(uintptr_t)(&As[b][srow * 32 + sc4]);
        ldsB[b] = (uint32_t)(uintptr_t)(&Bs[b][srow * 32 + sc4]);
    }
    const __bf16* xrow = X  + (size_t)(m0 + srow) * K + sc4;
    const __bf16* wrow = Wt + (size_t)(n0 + srow) * K + sc4;

    v8f acc[4];
    for (int j = 0; j < 4; ++j)
        for (int i = 0; i < 8; ++i) acc[j][i] = 0.0f;

    const int nt = K >> 5;
    // Prologue: stream tile 0 into buffer 0.
    async_copy_32B(ldsA[0], xrow);
    async_copy_32B(ldsB[0], wrow);

    for (int t = 0; t < nt; ++t) {
        const int b = t & 1;
        if (t + 1 < nt) {
            // Stream tile t+1 into the other buffer, then wait until only
            // those 4 async ops remain in flight (tile t is resident).
            async_copy_32B(ldsA[b ^ 1], xrow + (t + 1) * 32);
            async_copy_32B(ldsB[b ^ 1], wrow + (t + 1) * 32);
            if (t + 2 < nt) {  // warm L2 ahead of the async engine
                __builtin_prefetch(xrow + (t + 2) * 32, 0, 1);
                __builtin_prefetch(wrow + (t + 2) * 32, 0, 1);
            }
            asm volatile("s_wait_asynccnt 0x4" ::: "memory");
        } else {
            asm volatile("s_wait_asynccnt 0x0" ::: "memory");
        }
        __syncthreads();  // all waves' tile-t chunks resident

        // Pre-load all fragments into distinct registers so the scheduler
        // can drain DScnt incrementally across the WMMA sequence.
        v16bf afrag = load_a_frag(&As[b][(w * 16 + nr) * 32], kb);
        v16bf bfrag[4];
#pragma unroll
        for (int j = 0; j < 4; ++j)
            bfrag[j] = load_b_frag(&Bs[b][(16 * j + nr) * 32], grp);
#pragma unroll
        for (int j = 0; j < 4; ++j)
            acc[j] = __builtin_amdgcn_wmma_f32_16x16x32_bf16(
                false, afrag, false, bfrag[j], (short)0, acc[j], false, false);
        __syncthreads();  // all reads done before buffer b is re-streamed
    }

    // Epilogue. C layout: vgpr r, lane -> (row = r + 8*grp, col = 16*j + nr).
    const int rowbase = m0 + w * 16 + grp * 8;
    if (doRope) {
        // n0 is head-aligned: d = 16*j + nr; (d, d+32) sit in accumulators
        // j and j+2 of the same lane.
        for (int jp = 0; jp < 2; ++jp) {
            int d = 16 * jp + nr;
            for (int r = 0; r < 8; ++r) {
                int s = rowbase + r;
                float c  = cosT[s * HEADD + d];
                float sn = sinT[s * HEADD + d];
                float x1 = acc[jp][r], x2 = acc[jp + 2][r];
                acc[jp][r]     = x1 * c  - x2 * sn;
                acc[jp + 2][r] = x1 * sn + x2 * c;
            }
        }
    }
    for (int j = 0; j < 4; ++j) {
        for (int r = 0; r < 8; ++r) {
            int row = rowbase + r;
            int col = n0 + 16 * j + nr;
            float v = acc[j][r];
            if (outF) outF[(size_t)row * N + col] = v;
            if (outB) {
                __bf16 bb = (__bf16)(v * bscale);
                if (transB) outB[(size_t)col * M + row] = bb;  // V^T layout
                else        outB[(size_t)row * N + col] = bb;
            }
        }
    }
}

// ---------------------------------------------------------------------------
// Flash attention: one wave per (head, 16-row q tile). Causal, GQA 4:1.
// Q bf16 [s][2048] pre-scaled by 1/sqrt(64), K bf16 [s][512] post-RoPE,
// V^T bf16 [512][s]. Output bf16 [s][2048].
// ---------------------------------------------------------------------------
__global__ __launch_bounds__(32) void attn_fwd(
    const __bf16* __restrict__ Q, const __bf16* __restrict__ Kb,
    const __bf16* __restrict__ VT, __bf16* __restrict__ O, int S) {
    __shared__ __bf16 Ps[16 * 32];  // P staging: C-layout -> A-layout

    const int h    = blockIdx.x >> 7;
    const int t    = blockIdx.x & 127;
    const int q0   = t * 16;
    const int kv   = h >> 2;  // N_REP = 4
    const int lane = threadIdx.x;
    const int nr   = lane & 15;
    const int grp  = lane >> 4;
    const int kb   = grp * 8;

    // Q fragments: 16x64 -> two 16x32 A-frags, loaded once (b128 loads).
    const __bf16* qrow = Q + (size_t)(q0 + nr) * DMODEL + h * HEADD;
    v16bf qfrag[2];
    for (int f = 0; f < 2; ++f) qfrag[f] = load_a_frag(qrow + 32 * f, kb);

    v8f acc[4];
    for (int j = 0; j < 4; ++j)
        for (int i = 0; i < 8; ++i) acc[j][i] = 0.0f;
    float mrow[8], lrow[8];
    for (int r = 0; r < 8; ++r) { mrow[r] = -3e38f; lrow[r] = 0.0f; }

    const int kend = q0 + 16;
    for (int k0 = 0; k0 < kend; k0 += 32) {
        // ---- scores: two 16x16 tiles over 32 keys ----
        // Pre-load all 4 K fragments, then issue the 4 WMMAs.
        v16bf kfrag[4];
#pragma unroll
        for (int tb = 0; tb < 2; ++tb) {
            int srow = k0 + 16 * tb + nr;
            if (srow > S - 1) srow = S - 1;  // clamp; masked below
            const __bf16* krow =
                Kb + (size_t)srow * (NKVH * HEADD) + kv * HEADD;
            for (int f = 0; f < 2; ++f)
                kfrag[tb * 2 + f] = load_b_frag(krow + 32 * f, grp);
        }
        v8f sc[2];
#pragma unroll
        for (int tb = 0; tb < 2; ++tb) {
            for (int i = 0; i < 8; ++i) sc[tb][i] = 0.0f;
            for (int f = 0; f < 2; ++f)
                sc[tb] = __builtin_amdgcn_wmma_f32_16x16x32_bf16(
                    false, qfrag[f], false, kfrag[tb * 2 + f], (short)0,
                    sc[tb], false, false);
        }

        // ---- causal mask + online softmax (16-lane row reductions) ----
        for (int r = 0; r < 8; ++r) {
            int qi = q0 + grp * 8 + r;
            float s0 = sc[0][r]; if (k0 + nr      > qi) s0 = -1e30f;
            float s1 = sc[1][r]; if (k0 + 16 + nr > qi) s1 = -1e30f;
            float v = fmaxf(s0, s1);
            for (int mm = 1; mm < 16; mm <<= 1)
                v = fmaxf(v, __shfl_xor(v, mm, 32));
            float mn    = fmaxf(mrow[r], v);
            float alpha = __expf(mrow[r] - mn);
            float p0    = __expf(s0 - mn);
            float p1    = __expf(s1 - mn);
            float ps    = p0 + p1;
            for (int mm = 1; mm < 16; mm <<= 1)
                ps += __shfl_xor(ps, mm, 32);
            lrow[r] = lrow[r] * alpha + ps;
            mrow[r] = mn;
            for (int j = 0; j < 4; ++j) acc[j][r] *= alpha;
            Ps[(grp * 8 + r) * 32 + nr]      = (__bf16)p0;
            Ps[(grp * 8 + r) * 32 + 16 + nr] = (__bf16)p1;
        }
        asm volatile("s_wait_dscnt 0x0" ::: "memory");

        // ---- reload P as 16x32 A fragment ----
        v16bf pfrag = load_a_frag(Ps + nr * 32, kb);

        // ---- O += P @ V (V^T gives K-contiguous b128 loads) ----
        int kbase = k0 + grp * 16;
        if (kbase > S - 16) kbase = S - 16;  // chunk clamp (P==0 there)
        v16bf vfrag[4];
#pragma unroll
        for (int j = 0; j < 4; ++j) {
            const __bf16* vcol =
                VT + (size_t)(kv * HEADD + 16 * j + nr) * S + kbase;
            v4u lo = *(const v4u*)(vcol);
            v4u hi = *(const v4u*)(vcol + 8);
            vfrag[j] = combine_bf(lo, hi);
        }
#pragma unroll
        for (int j = 0; j < 4; ++j)
            acc[j] = __builtin_amdgcn_wmma_f32_16x16x32_bf16(
                false, pfrag, false, vfrag[j], (short)0, acc[j], false, false);
        asm volatile("s_wait_dscnt 0x0" ::: "memory");
    }

    // ---- normalize and store ----
    for (int j = 0; j < 4; ++j)
        for (int r = 0; r < 8; ++r) {
            float o = acc[j][r] / lrow[r];
            O[(size_t)(q0 + grp * 8 + r) * DMODEL + h * HEADD + 16 * j + nr] =
                (__bf16)o;
        }
}

// ---------------------------------------------------------------------------
// Host-side orchestration
// ---------------------------------------------------------------------------
extern "C" void kernel_launch(void* const* d_in, const int* in_sizes, int n_in,
                              void* d_out, int out_size, void* d_ws,
                              size_t ws_size, hipStream_t stream) {
    const float* x    = (const float*)d_in[0];
    const float* cosT = (const float*)d_in[1];
    const float* sinT = (const float*)d_in[2];
    // d_in[3] = mask: causal mask applied analytically in attn_fwd.
    const float* wq = (const float*)d_in[4];
    const float* wk = (const float*)d_in[5];
    const float* wv = (const float*)d_in[6];
    const float* wo = (const float*)d_in[7];

    // Workspace layout (bytes). Weight buffers hold TRANSPOSED bf16 [N][K].
    char* ws = (char*)d_ws;
    __bf16* xb   = (__bf16*)(ws + 0);          // x bf16        [2048][2048]
    __bf16* wqT  = (__bf16*)(ws + 8388608);    // wq^T bf16     [2048][2048]
    __bf16* wkT  = (__bf16*)(ws + 16777216);   // wk^T bf16     [512][2048]
    __bf16* wvT  = (__bf16*)(ws + 18874368);   // wv^T bf16     [512][2048]
    __bf16* woT  = (__bf16*)(ws + 20971520);   // wo^T bf16     [2048][2048]
    __bf16* qb   = (__bf16*)(ws + 29360128);   // q roped*0.125 [2048][2048]
    __bf16* kbf  = (__bf16*)(ws + 37748736);   // k roped       [2048][512]
    __bf16* vtb  = (__bf16*)(ws + 39845888);   // v^T           [512][2048]
    __bf16* ab   = (__bf16*)(ws + 41943040);   // attn out      [2048][2048]

    float* out = (float*)d_out;
    float* nk  = out + (size_t)S_LEN * DMODEL;       // new_k fp32
    float* nv  = nk + (size_t)S_LEN * NKVH * HEADD;  // new_v fp32

    // 1) conversions (weights transposed so GEMM staging is fully coalesced)
    cvt_f32_bf16<<<16384, 256, 0, stream>>>(x, xb, S_LEN * DMODEL);
    cvt_transpose_bf16<<<dim3(64, 64), dim3(32, 8), 0, stream>>>(
        wq, wqT, DMODEL, DMODEL);
    cvt_transpose_bf16<<<dim3(16, 64), dim3(32, 8), 0, stream>>>(
        wk, wkT, DMODEL, NKVH * HEADD);
    cvt_transpose_bf16<<<dim3(16, 64), dim3(32, 8), 0, stream>>>(
        wv, wvT, DMODEL, NKVH * HEADD);
    cvt_transpose_bf16<<<dim3(64, 64), dim3(32, 8), 0, stream>>>(
        wo, woT, DMODEL, DMODEL);

    // 2) Q = rope(x@wq) * 0.125 -> bf16
    gemm_bf16<<<dim3(32, 32), 128, 0, stream>>>(
        xb, wqT, nullptr, qb, cosT, sinT, S_LEN, DMODEL, DMODEL, 1, 0, 0.125f);
    // 3) K = rope(x@wk) -> bf16 + fp32 new_k
    gemm_bf16<<<dim3(8, 32), 128, 0, stream>>>(
        xb, wkT, nk, kbf, cosT, sinT, S_LEN, NKVH * HEADD, DMODEL, 1, 0, 1.0f);
    // 4) V = x@wv -> bf16 transposed + fp32 new_v
    gemm_bf16<<<dim3(8, 32), 128, 0, stream>>>(
        xb, wvT, nv, vtb, nullptr, nullptr, S_LEN, NKVH * HEADD, DMODEL,
        0, 1, 1.0f);

    // 5) flash attention: 32 heads x 128 q-tiles, one wave each
    attn_fwd<<<NHEAD * (S_LEN / 16), 32, 0, stream>>>(qb, kbf, vtb, ab, S_LEN);

    // 6) out = attn @ wo -> fp32
    gemm_bf16<<<dim3(32, 32), 128, 0, stream>>>(
        ab, woT, out, nullptr, nullptr, nullptr, S_LEN, DMODEL, DMODEL,
        0, 0, 1.0f);
}
